// BaseTransformer_25288767438913
// MI455X (gfx1250) — matrix operand order repaired and checked by reading, not compiled, not run on hardware
//
#include <hip/hip_runtime.h>
#include <cstdint>
#include <cstddef>

// ---------------------------------------------------------------------------
// MI455X / gfx1250 transformer forward.  bf16 WMMA (v_wmma_f32_16x16x32_bf16)
// with f32 accumulation; flash-style attention (row-sums via WMMA on a
// ones-matrix, row-max via ds_swizzle xor butterflies); tiled GEMMs with
// async global->LDS staging (ASYNCcnt) for transpose-free tiles and a
// register pipeline + LDS transpose for B; global_prefetch look-ahead.
// ---------------------------------------------------------------------------

typedef __attribute__((ext_vector_type(16))) __bf16        v16bf;
typedef __attribute__((ext_vector_type(8)))  float         v8f;
typedef __attribute__((ext_vector_type(4)))  unsigned int  u32x4;
typedef __attribute__((ext_vector_type(4)))  float         f32x4;

#define TDIM   1024
#define THEADS 16
#define TDH    64
#define TSEQ   1024
#define TBATCH 2
#define TROWS  (TBATCH * TSEQ)   // 2048
#define TFF    4096
#define TDEPTH 6

__device__ __forceinline__ unsigned short f2bf(float f) {
  unsigned int u = __float_as_uint(f);
  u += 0x7FFFu + ((u >> 16) & 1u);           // round-to-nearest-even
  return (unsigned short)(u >> 16);
}

// ds_swizzle_b32 xor-mode: offset = {1'b0, xor[4:0], or[4:0]=0, and[4:0]=0x1f}
#define SWZ_XOR(x, m) \
  __uint_as_float(__builtin_amdgcn_ds_swizzle(__float_as_uint(x), (((m) << 10) | 0x1f)))

// Async global->LDS copy (16B per lane).  LDS byte offset = low 32 bits of the
// flat address of a __shared__ element (aperture lives in the high 32 bits).
__device__ __forceinline__ void async_ld_b128(const unsigned short* lds_elt,
                                              const void* gaddr) {
  unsigned lds_off = (unsigned)(uintptr_t)lds_elt;
  asm volatile("global_load_async_to_lds_b128 %0, %1, off"
               :: "v"(lds_off), "v"(gaddr) : "memory");
}
__device__ __forceinline__ void wait_async0() {
  asm volatile("s_wait_asynccnt 0x0" ::: "memory");
}

union Frag { v16bf bf; u32x4 q[2]; unsigned short h[16]; };
union Acc  { v8f v; float f[8]; };

__device__ __forceinline__ v8f wmma_bf16(const Frag& a, const Frag& b, v8f c) {
  return __builtin_amdgcn_wmma_f32_16x16x32_bf16(
      /*neg_a=*/false, a.bf, /*neg_b=*/false, b.bf,
      /*c_mod=*/(short)0, c, /*reuse_a=*/false, /*reuse_b=*/false);
}

// ---------------------------------------------------------------------------
// f32 -> bf16 bulk conversion (weights)
// ---------------------------------------------------------------------------
__global__ void __launch_bounds__(256)
k_f32_to_bf16(const float* __restrict__ in, unsigned short* __restrict__ out, int n) {
  int i = (blockIdx.x * 256 + threadIdx.x) * 4;
  if (i + 3 < n) {
    f32x4 v = *(const f32x4*)(in + i);
    unsigned int p0 = (unsigned int)f2bf(v.x) | ((unsigned int)f2bf(v.y) << 16);
    unsigned int p1 = (unsigned int)f2bf(v.z) | ((unsigned int)f2bf(v.w) << 16);
    *(unsigned int*)(out + i)     = p0;
    *(unsigned int*)(out + i + 2) = p1;
  }
}

// ---------------------------------------------------------------------------
// LayerNorm (one 1024-wide row per block) -> bf16
// ---------------------------------------------------------------------------
__global__ void __launch_bounds__(256)
k_layernorm_bf16(const float* __restrict__ x, const float* __restrict__ g,
                 const float* __restrict__ b, unsigned short* __restrict__ out) {
  __shared__ float red[256];
  const int row = blockIdx.x;
  const int t = threadIdx.x;
  const float* xr = x + (size_t)row * TDIM;

  float v[4];
  float s = 0.f;
#pragma unroll
  for (int i = 0; i < 4; ++i) { v[i] = xr[t + i * 256]; s += v[i]; }
  red[t] = s; __syncthreads();
  for (int off = 128; off > 0; off >>= 1) {
    if (t < off) red[t] += red[t + off];
    __syncthreads();
  }
  const float mu = red[0] * (1.f / TDIM);
  __syncthreads();
  float s2 = 0.f;
#pragma unroll
  for (int i = 0; i < 4; ++i) { float d = v[i] - mu; s2 += d * d; }
  red[t] = s2; __syncthreads();
  for (int off = 128; off > 0; off >>= 1) {
    if (t < off) red[t] += red[t + off];
    __syncthreads();
  }
  const float rs = rsqrtf(red[0] * (1.f / TDIM) + 1e-5f);
#pragma unroll
  for (int i = 0; i < 4; ++i) {
    int idx = t + i * 256;
    out[(size_t)row * TDIM + idx] = f2bf((v[i] - mu) * rs * g[idx] + b[idx]);
  }
}

// ---------------------------------------------------------------------------
// Tiled bf16 WMMA GEMM:  D(MxN) = A(MxK) * B(KxN)  [+ epilogue]
//   block tile 128x256, K-step 32, 8 waves (2x4), each wave 64x64 (4x4 WMMAs)
//   A tile: async global->LDS (double-buffered); B tile: register pipeline +
//   LDS transpose; global_prefetch for the tile after next.
//   MODE 0: split into Q/K/V head-major bf16 (q scaled)   -> outH = qkv base
//   MODE 1: outF[m][N] += D + bias   (residual, f32)
//   MODE 2: outH[m][N]  = bf16(gelu(D + bias))
// ---------------------------------------------------------------------------
#define BM 128
#define BN 256
#define BK 32
#define LDA 40   // padded row stride (halfs): 80B, 16B aligned, bank friendly
#define LDBT 40

template <int MODE>
__global__ void __launch_bounds__(256)
k_gemm_bf16(const unsigned short* __restrict__ A,   // [M][K] bf16
            const unsigned short* __restrict__ B,   // [K][N] bf16
            float* __restrict__ outF,
            unsigned short* __restrict__ outH,
            const float* __restrict__ bias,
            int M, int N, int K, float qscale) {
  __shared__ unsigned short As[2][BM * LDA];  // 2 x 10.0 KB (async target)
  __shared__ unsigned short Bt[BN * LDBT];    // 20.0 KB  (stored [n][k])

  const int tid  = threadIdx.x;
  const int lane = tid & 31;
  const int wave = tid >> 5;
  const int wm = wave >> 2;          // 0..1
  const int wn = wave & 3;           // 0..3
  const int hi = (lane >> 4) & 1;    // which 16-lane half
  const int ln = lane & 15;

  const int bm = blockIdx.y;
  const int bn = blockIdx.x;

  // per-thread static staging positions
  int ar[2], ac[2];
  const unsigned short* aptr[2];
#pragma unroll
  for (int c = 0; c < 2; ++c) {
    int e = (tid + c * 256) * 8;
    ar[c] = e >> 5; ac[c] = e & 31;
    aptr[c] = A + (size_t)(bm * BM + ar[c]) * K + ac[c];
  }
  int bk_[4], bn_[4];
  const unsigned short* bptr[4];
#pragma unroll
  for (int c = 0; c < 4; ++c) {
    int e = (tid + c * 256) * 8;
    bk_[c] = e >> 8; bn_[c] = e & 255;
    bptr[c] = B + (size_t)bk_[c] * N + bn * BN + bn_[c];
  }

  Acc acc[4][4];
#pragma unroll
  for (int i = 0; i < 4; ++i)
#pragma unroll
    for (int j = 0; j < 4; ++j)
#pragma unroll
      for (int r = 0; r < 8; ++r) acc[i][j].f[r] = 0.f;

  // prologue: async A tile 0 -> As[0]; B tile 0 -> VGPRs
#pragma unroll
  for (int c = 0; c < 2; ++c)
    async_ld_b128(&As[0][ar[c] * LDA + ac[c]], aptr[c]);
  u32x4 bReg[4];
#pragma unroll
  for (int c = 0; c < 4; ++c) bReg[c] = *(const u32x4*)(bptr[c]);

  int buf = 0;
  for (int k0 = 0; k0 < K; k0 += BK) {
    // ---- commit staged B (transposed -> Bt[n][k]) ----
#pragma unroll
    for (int c = 0; c < 4; ++c) {
      const unsigned short* ds = (const unsigned short*)&bReg[c];
#pragma unroll
      for (int i = 0; i < 8; ++i) Bt[(bn_[c] + i) * LDBT + bk_[c]] = ds[i];
    }
    wait_async0();            // A tile for this iteration landed in As[buf]
    __syncthreads();

    // ---- kick off next tile (async A, register B), overlap with WMMA ----
    if (k0 + BK < K) {
#pragma unroll
      for (int c = 0; c < 2; ++c)
        async_ld_b128(&As[buf ^ 1][ar[c] * LDA + ac[c]], aptr[c] + (k0 + BK));
#pragma unroll
      for (int c = 0; c < 4; ++c)
        bReg[c] = *(const u32x4*)(bptr[c] + (size_t)(k0 + BK) * N);
      if (k0 + 2 * BK < K) {
        __builtin_prefetch(aptr[0] + (k0 + 2 * BK), 0, 1);
        __builtin_prefetch(bptr[0] + (size_t)(k0 + 2 * BK) * N, 0, 1);
      }
    }

    // ---- fragments per the CDNA5 16-bit A / B VGPR layouts ----
    const unsigned short* Ab = &As[buf][0];
    Frag af[4], bf[4];
#pragma unroll
    for (int mi = 0; mi < 4; ++mi) {
      int row = wm * 64 + mi * 16 + ln;
      int kb = hi * 8;   // lanes>=16 hold k=8..15 / 24..31
      af[mi].q[0] = *(const u32x4*)(&Ab[row * LDA + kb]);
      af[mi].q[1] = *(const u32x4*)(&Ab[row * LDA + kb + 16]);
    }
#pragma unroll
    for (int ni = 0; ni < 4; ++ni) {
      int n = wn * 64 + ni * 16 + ln;
      int kb = hi * 8;   // lanes>=16 hold k=8..15 / 24..31
      bf[ni].q[0] = *(const u32x4*)(&Bt[n * LDBT + kb]);
      bf[ni].q[1] = *(const u32x4*)(&Bt[n * LDBT + kb + 16]);
    }
#pragma unroll
    for (int mi = 0; mi < 4; ++mi)
#pragma unroll
      for (int ni = 0; ni < 4; ++ni)
        acc[mi][ni].v = wmma_bf16(af[mi], bf[ni], acc[mi][ni].v);
    __syncthreads();
    buf ^= 1;
  }

  // ---- epilogue (C layout: row = r + 8*hi, col = lane&15) ----
#pragma unroll
  for (int mi = 0; mi < 4; ++mi)
#pragma unroll
    for (int ni = 0; ni < 4; ++ni)
#pragma unroll
      for (int r = 0; r < 8; ++r) {
        int grow = bm * BM + wm * 64 + mi * 16 + r + hi * 8;
        int gcol = bn * BN + wn * 64 + ni * 16 + ln;
        float val = acc[mi][ni].f[r];
        if (MODE == 0) {
          int sec = gcol >> 10;            // 0=q 1=k 2=v
          int cin = gcol & 1023;
          int head = cin >> 6, dh = cin & 63;
          int bb = grow >> 10, nn = grow & 1023;
          if (sec == 0) val *= qscale;
          outH[(size_t)sec * (TBATCH * THEADS * TSEQ * TDH) +
               (((size_t)(bb * THEADS + head) * TSEQ + nn) * TDH) + dh] = f2bf(val);
        } else if (MODE == 1) {
          size_t o = (size_t)grow * N + gcol;
          outF[o] += val + bias[gcol];
        } else {
          float v2 = val + bias[gcol];
          float ge = 0.5f * v2 * (1.f + erff(v2 * 0.70710678118f));
          outH[(size_t)grow * N + gcol] = f2bf(ge);
        }
      }
}

// ---------------------------------------------------------------------------
// Flash attention: grid (SEQ/128, B*H); 8 waves x 16 queries per block.
// 32-key tiles; K tile async global->LDS (double-buffered), V tile via
// register pipeline + LDS transpose; online softmax: row-max via ds_swizzle
// xor butterfly, row-sum via WMMA against a ones-matrix; P bounced via LDS.
// Writes o as bf16 in [b, n, h*64+dh] layout (GEMM-ready).
// ---------------------------------------------------------------------------
__global__ void __launch_bounds__(256)
k_attention(const unsigned short* __restrict__ qkv,   // [3][B*H][SEQ][DH] bf16
            const float* __restrict__ bias,           // [H][SEQ][SEQ] f32
            unsigned short* __restrict__ o) {         // [B][SEQ][DIM] bf16
  __shared__ unsigned short Kt[2][32 * 72];     // [key][dh]  2 x 4.5 KB (async)
  __shared__ unsigned short Vt[64 * 40];        // [dh][key]  5.0 KB (transposed)
  __shared__ unsigned short Pb[8 * 16 * 40];    // per-wave P 16x32, 10 KB

  const int tid = threadIdx.x, lane = tid & 31, wave = tid >> 5;
  const int hi = (lane >> 4) & 1, ln = lane & 15;
  const int bh = blockIdx.y;
  const int b = bh >> 4, h = bh & 15;
  const int qbase = blockIdx.x * 128 + wave * 16;
  const size_t SEC = (size_t)TBATCH * THEADS * TSEQ * TDH;
  const unsigned short* qm = qkv + (size_t)bh * TSEQ * TDH;
  const unsigned short* km = qm + SEC;
  const unsigned short* vm = qm + 2 * SEC;

  // Q fragments (scale folded at QKV epilogue): A 16x32 layout, 2 K-steps
  Frag qf[2];
#pragma unroll
  for (int f = 0; f < 2; ++f) {
    int row = qbase + ln;
    int kb = f * 32 + hi * 8;
    qf[f].q[0] = *(const u32x4*)(qm + (size_t)row * TDH + kb);
    qf[f].q[1] = *(const u32x4*)(qm + (size_t)row * TDH + kb + 16);
  }

  // ones B-matrix (bf16 1.0) for row-sum WMMA
  Frag ones;
#pragma unroll
  for (int i = 0; i < 16; ++i) ones.h[i] = 0x3F80;

  Acc oacc[4];
#pragma unroll
  for (int t = 0; t < 4; ++t)
#pragma unroll
    for (int r = 0; r < 8; ++r) oacc[t].f[r] = 0.f;
  float mrow[8], lrow[8];
#pragma unroll
  for (int r = 0; r < 8; ++r) { mrow[r] = -1e30f; lrow[r] = 0.f; }

  unsigned short* Pw = &Pb[wave * 16 * 40];

  // per-thread staging position + prologue (async K tile 0, V tile 0 in regs)
  const int skey = (tid * 8) >> 6, sdh = (tid * 8) & 63;
  const unsigned short* kbase = km + (size_t)skey * TDH + sdh;
  const unsigned short* vbase = vm + (size_t)skey * TDH + sdh;
  async_ld_b128(&Kt[0][skey * 72 + sdh], kbase);
  u32x4 vReg = *(const u32x4*)(vbase);

  int buf = 0;
  for (int jb = 0; jb < TSEQ; jb += 32) {
    // ---- commit staged V (transposed); K landed via async ----
    {
      const unsigned short* vs = (const unsigned short*)&vReg;
#pragma unroll
      for (int i = 0; i < 8; ++i) Vt[(sdh + i) * 40 + skey] = vs[i];
    }
    wait_async0();
    __syncthreads();

    // ---- kick off next tile (async K, register V) ----
    if (jb + 32 < TSEQ) {
      async_ld_b128(&Kt[buf ^ 1][skey * 72 + sdh],
                    kbase + (size_t)(jb + 32) * TDH);
      vReg = *(const u32x4*)(vbase + (size_t)(jb + 32) * TDH);
    }

    // ---- S = Q x K^T  (two 16x16 tiles over 32 keys) ----
    const unsigned short* Kb = &Kt[buf][0];
    Acc s0, s1;
#pragma unroll
    for (int r = 0; r < 8; ++r) { s0.f[r] = 0.f; s1.f[r] = 0.f; }
#pragma unroll
    for (int ks = 0; ks < 2; ++ks) {
      Frag kb0, kb1;
      int kbo = ks * 32 + hi * 8;         // B frag: k = dh, n = key
      kb0.q[0] = *(const u32x4*)(&Kb[ln * 72 + kbo]);
      kb0.q[1] = *(const u32x4*)(&Kb[ln * 72 + kbo + 16]);
      kb1.q[0] = *(const u32x4*)(&Kb[(16 + ln) * 72 + kbo]);
      kb1.q[1] = *(const u32x4*)(&Kb[(16 + ln) * 72 + kbo + 16]);
      s0.v = wmma_bf16(qf[ks], kb0, s0.v);
      s1.v = wmma_bf16(qf[ks], kb1, s1.v);
    }

    // ---- bias add + online softmax max path (ds_swizzle butterflies) ----
    float corr[8];
#pragma unroll
    for (int r = 0; r < 8; ++r) {
      int qr = qbase + r + hi * 8;
      const float* brow = bias + ((size_t)h * TSEQ + qr) * TSEQ + jb;
      float a0 = s0.f[r] + brow[ln];
      float a1 = s1.f[r] + brow[16 + ln];
      float mx = fmaxf(a0, a1);
      mx = fmaxf(mx, SWZ_XOR(mx, 1));
      mx = fmaxf(mx, SWZ_XOR(mx, 2));
      mx = fmaxf(mx, SWZ_XOR(mx, 4));
      mx = fmaxf(mx, SWZ_XOR(mx, 8));
      float mnew = fmaxf(mrow[r], mx);
      float e0 = __expf(a0 - mnew), e1 = __expf(a1 - mnew);
      corr[r] = __expf(mrow[r] - mnew);
      mrow[r] = mnew;
      Pw[(r + hi * 8) * 40 + ln]      = f2bf(e0);
      Pw[(r + hi * 8) * 40 + 16 + ln] = f2bf(e1);
    }

    // ---- P A-fragment (via LDS re-layout), row sums via P x ones ----
    Frag pa;
    {
      int kb2 = hi * 8;
      pa.q[0] = *(const u32x4*)(&Pw[ln * 40 + kb2]);
      pa.q[1] = *(const u32x4*)(&Pw[ln * 40 + kb2 + 16]);
    }
    Acc rsum;
#pragma unroll
    for (int r = 0; r < 8; ++r) rsum.f[r] = 0.f;
    rsum.v = wmma_bf16(pa, ones, rsum.v);   // every lane: row sum of P
#pragma unroll
    for (int r = 0; r < 8; ++r) {
      lrow[r] = lrow[r] * corr[r] + rsum.f[r];
#pragma unroll
      for (int t = 0; t < 4; ++t) oacc[t].f[r] *= corr[r];
    }

    // ---- O += P x V (V from transposed LDS) ----
#pragma unroll
    for (int t = 0; t < 4; ++t) {
      Frag vb;
      int kb2 = hi * 8;                   // B frag: k = key, n = dh
      vb.q[0] = *(const u32x4*)(&Vt[(t * 16 + ln) * 40 + kb2]);
      vb.q[1] = *(const u32x4*)(&Vt[(t * 16 + ln) * 40 + kb2 + 16]);
      oacc[t].v = wmma_bf16(pa, vb, oacc[t].v);
    }
    __syncthreads();
    buf ^= 1;
  }

  // ---- normalize + store o ----
#pragma unroll
  for (int r = 0; r < 8; ++r) {
    float inv = 1.f / lrow[r];
    int qr = qbase + r + hi * 8;
#pragma unroll
    for (int t = 0; t < 4; ++t)
      o[((size_t)(b * TSEQ + qr)) * TDIM + h * TDH + t * 16 + ln] =
          f2bf(oacc[t].f[r] * inv);
  }
}

// ---------------------------------------------------------------------------
// Host orchestration.  Workspace layout (needs >= 40 MB):
//   [0,8MB)   WB   bf16 weights (reused per GEMM)
//   [8,12MB)  Abf  bf16 activations [2048x1024]
//   [12,24MB) QKV  bf16 [3][B*H][SEQ][DH]
//   [24,40MB) Hbf  bf16 FFN hidden [2048x4096]
// Running x lives in d_out (f32, residual updated in place).
// ---------------------------------------------------------------------------
extern "C" void kernel_launch(void* const* d_in, const int* in_sizes, int n_in,
                              void* d_out, int out_size, void* d_ws, size_t ws_size,
                              hipStream_t stream) {
  const float* x    = (const float*)d_in[0];
  const float* bias = (const float*)d_in[1];
  const float* ln1g = (const float*)d_in[2];
  const float* ln1b = (const float*)d_in[3];
  const float* wqkv = (const float*)d_in[4];
  const float* wout = (const float*)d_in[5];
  const float* bout = (const float*)d_in[6];
  const float* ln2g = (const float*)d_in[7];
  const float* ln2b = (const float*)d_in[8];
  const float* w1   = (const float*)d_in[9];
  const float* b1   = (const float*)d_in[10];
  const float* w2   = (const float*)d_in[11];
  const float* b2   = (const float*)d_in[12];
  float* xbuf = (float*)d_out;

  char* ws = (char*)d_ws;
  unsigned short* WB  = (unsigned short*)(ws);
  unsigned short* Abf = (unsigned short*)(ws + ((size_t)8  << 20));
  unsigned short* QKV = (unsigned short*)(ws + ((size_t)12 << 20));
  unsigned short* Hbf = (unsigned short*)(ws + ((size_t)24 << 20));

  hipMemcpyAsync(xbuf, x, sizeof(float) * (size_t)TROWS * TDIM,
                 hipMemcpyDeviceToDevice, stream);

  const float qscale = 0.125f;  // 1/sqrt(dh)

  for (int l = 0; l < TDEPTH; ++l) {
    // ---- attention block ----
    k_layernorm_bf16<<<TROWS, 256, 0, stream>>>(xbuf, ln1g + l * TDIM,
                                                ln1b + l * TDIM, Abf);
    k_f32_to_bf16<<<(TDIM * 3 * TDIM) / 1024, 256, 0, stream>>>(
        wqkv + (size_t)l * TDIM * 3 * TDIM, WB, TDIM * 3 * TDIM);
    k_gemm_bf16<0><<<dim3(3 * TDIM / BN, TROWS / BM), 256, 0, stream>>>(
        Abf, WB, nullptr, QKV, nullptr, TROWS, 3 * TDIM, TDIM, qscale);
    k_attention<<<dim3(TSEQ / 128, TBATCH * THEADS), 256, 0, stream>>>(
        QKV, bias, Abf);
    k_f32_to_bf16<<<(TDIM * TDIM) / 1024, 256, 0, stream>>>(
        wout + (size_t)l * TDIM * TDIM, WB, TDIM * TDIM);
    k_gemm_bf16<1><<<dim3(TDIM / BN, TROWS / BM), 256, 0, stream>>>(
        Abf, WB, xbuf, nullptr, bout + l * TDIM, TROWS, TDIM, TDIM, 1.f);

    // ---- FFN block ----
    k_layernorm_bf16<<<TROWS, 256, 0, stream>>>(xbuf, ln2g + l * TDIM,
                                                ln2b + l * TDIM, Abf);
    k_f32_to_bf16<<<(TDIM * TFF) / 1024, 256, 0, stream>>>(
        w1 + (size_t)l * TDIM * TFF, WB, TDIM * TFF);
    k_gemm_bf16<2><<<dim3(TFF / BN, TROWS / BM), 256, 0, stream>>>(
        Abf, WB, nullptr, Hbf, b1 + l * TFF, TROWS, TFF, TDIM, 1.f);
    k_f32_to_bf16<<<(TFF * TDIM) / 1024, 256, 0, stream>>>(
        w2 + (size_t)l * TFF * TDIM, WB, TFF * TDIM);
    k_gemm_bf16<1><<<dim3(TDIM / BN, TROWS / BM), 256, 0, stream>>>(
        Hbf, WB, xbuf, nullptr, b2 + l * TDIM, TROWS, TDIM, TFF, 1.f);
  }
}
